// AdditiveAttention_52656299049100
// MI455X (gfx1250) — matrix operand order repaired and checked by reading, not compiled, not run on hardware
//
#include <hip/hip_runtime.h>
#include <math.h>

#define B_    32
#define S_    2048
#define D_    512          // VD = MD = AD
#define ROWS  128          // s-rows per workgroup
#define PITCH 520          // padded LDS row pitch (elements) -> bank-conflict friendly

typedef __attribute__((ext_vector_type(16))) __bf16         v16bf;
typedef __attribute__((ext_vector_type(8)))  float          v8f;
typedef __attribute__((ext_vector_type(4)))  unsigned short u16x4;
typedef __attribute__((ext_vector_type(8)))  unsigned short u16x8;
typedef __attribute__((ext_vector_type(16))) unsigned short u16x16;

// f32 -> bf16, round-to-nearest-even (inputs are finite normals)
__device__ __forceinline__ unsigned short f2bf(float f) {
  unsigned int u = __float_as_uint(f);
  u += 0x7FFFu + ((u >> 16) & 1u);
  return (unsigned short)(u >> 16);
}

__device__ __forceinline__ void store_bf16x4(unsigned short* dst, float4 f) {
  u16x4 p;
  p.x = f2bf(f.x); p.y = f2bf(f.y); p.z = f2bf(f.z); p.w = f2bf(f.w);
  *(u16x4*)dst = p;   // single 8-byte ds_store
}

// Branch-free tanh (native v_tanh_f32 confirmed on gfx1250)
__device__ __forceinline__ float tanh_fast(float x) {
#if __has_builtin(__builtin_amdgcn_tanhf)
  return __builtin_amdgcn_tanhf(x);
#else
  float e = __expf(2.0f * x);
  return fmaf(-2.0f, __builtin_amdgcn_rcpf(e + 1.0f), 1.0f);
#endif
}

// Assemble a v16bf WMMA fragment from two 16-byte LDS loads.
__device__ __forceinline__ v16bf lds_frag(const unsigned short* p0, const unsigned short* p1) {
  u16x8 lo = *(const u16x8*)p0;
  u16x8 hi = *(const u16x8*)p1;
  u16x16 t = __builtin_shufflevector(lo, hi, 0,1,2,3,4,5,6,7,8,9,10,11,12,13,14,15);
  return __builtin_bit_cast(v16bf, t);
}

// ---------------- Kernel 1: w1q[b][a] = sum_d query[b][d] * W1[a][d] ----------------
__global__ __launch_bounds__(256) void w1q_kernel(const float* __restrict__ query,
                                                  const float* __restrict__ W1,
                                                  float* __restrict__ w1q) {
  const int idx = blockIdx.x * 256 + threadIdx.x;  // b*512 + a
  const int b = idx >> 9;
  const int a = idx & 511;
  const float4* qp = (const float4*)(query + (size_t)b * D_);
  const float4* wp = (const float4*)(W1 + (size_t)a * D_);
  float s = 0.f;
#pragma unroll 4
  for (int i = 0; i < D_ / 4; ++i) {
    float4 x = qp[i], y = wp[i];
    s = fmaf(x.x, y.x, s);
    s = fmaf(x.y, y.y, s);
    s = fmaf(x.z, y.z, s);
    s = fmaf(x.w, y.w, s);
  }
  w1q[idx] = s;
}

// ---------------- Kernel 2: scores[b][s] = sum_a tanh(w1q[b][a] + keys[b][s]·W2[a]) * v[a] ----
// LDS layout puts the B slab at offset 0 so every ds_load in the hot loop is
// base-VGPR + 16-bit immediate (no per-iteration address VALU).
__global__ __attribute__((amdgpu_waves_per_eu(2, 2))) __launch_bounds__(256)
void att_scores_kernel(const float* __restrict__ keys,
                       const float* __restrict__ W2,
                       const float* __restrict__ w1q,
                       const float* __restrict__ vvec,
                       float* __restrict__ scores) {
  extern __shared__ __align__(16) unsigned short smem_u16[];
  unsigned short* bS    = smem_u16;               // 16 * PITCH bf16   (offset 0)
  float* w1qS           = (float*)(bS + 16 * PITCH);   // D_ floats
  float* vS             = w1qS + D_;                   // D_ floats
  unsigned short* keysS = (unsigned short*)(vS + D_);  // ROWS * PITCH bf16

  const int sblk = blockIdx.x;        // 0..S_/ROWS-1
  const int b    = blockIdx.y;        // 0..B_-1
  const int tid  = threadIdx.x;
  const int lane = tid & 31;
  const int wid  = tid >> 5;          // 8 waves, one 16-row tile each
  const int s0   = sblk * ROWS;
  const int l16  = lane & 15;
  const int hsel = lane >> 4;         // lane-half select (ISA 16-bit fragment layout)

  // Stage w1q row for this batch + v vector.
  for (int i = tid; i < D_; i += 256) {
    w1qS[i] = w1q[(size_t)b * D_ + i];
    vS[i]   = vvec[i];
  }

  // Stage keys[b, s0:s0+ROWS, :] as bf16 into LDS (coalesced float4 reads).
  const float* kbase = keys + ((size_t)b * S_ + s0) * D_;
  for (int idx = tid; idx < ROWS * D_ / 4; idx += 256) {
    const int e = idx * 4;
    const int r = e >> 9;      // / 512
    const int c = e & 511;
    float4 f = *(const float4*)(kbase + (size_t)r * D_ + c);
    store_bf16x4(keysS + r * PITCH + c, f);
  }

  // Hoisted per-lane fragment bases (loop offsets become DS immediates).
  const int mrow = wid * 16 + l16;                       // A-fragment row
  const unsigned short* arow = keysS + mrow * PITCH + hsel * 16;
  const unsigned short* brow = bS + l16 * PITCH + hsel * 16;

  float sp[8];
#pragma unroll
  for (int r = 0; r < 8; ++r) sp[r] = 0.f;

  for (int nt = 0; nt < D_ / 16; ++nt) {
    __syncthreads();  // previous tile's B reads done (also covers keys staging)
    // Stage B slab: W2[nt*16 .. nt*16+15][0..511] -> bf16 (W2 is L2-resident)
    for (int idx = tid; idx < 16 * D_ / 4; idx += 256) {
      const int e = idx * 4;
      const int r = e >> 9;
      const int c = e & 511;
      float4 f = *(const float4*)(W2 + (size_t)(nt * 16 + r) * D_ + c);
      store_bf16x4(bS + r * PITCH + c, f);
    }
    __syncthreads();

    v8f c8 = {0.f, 0.f, 0.f, 0.f, 0.f, 0.f, 0.f, 0.f};
#pragma unroll 4
    for (int kk = 0; kk < D_ / 32; ++kk) {
      // A (16x32 bf16): lane = row m, 16 consecutive K per lane-half (K = kk*32 + hsel*16 + e)
      v16bf a  = lds_frag(arow + kk * 32, arow + kk * 32 + 8);
      // B (32x16 bf16): lane = column n, 16 consecutive K per lane-half
      v16bf bm = lds_frag(brow + kk * 32, brow + kk * 32 + 8);
      c8 = __builtin_amdgcn_wmma_f32_16x16x32_bf16(false, a, false, bm,
                                                   (short)0, c8, false, false);
    }

    // Epilogue: each lane holds column a = nt*16 + l16, rows hsel*8 + r.
    const int acol = nt * 16 + l16;
    const float w1v = w1qS[acol];
    const float vv  = vS[acol];
#pragma unroll
    for (int r = 0; r < 8; ++r)
      sp[r] += tanh_fast(c8[r] + w1v) * vv;
  }

  // Reduce over the 16 lanes (columns) that share each output row.
#pragma unroll
  for (int r = 0; r < 8; ++r) {
    float x = sp[r];
    x += __shfl_xor(x, 1, 32);
    x += __shfl_xor(x, 2, 32);
    x += __shfl_xor(x, 4, 32);
    x += __shfl_xor(x, 8, 32);
    sp[r] = x;
  }
  if (l16 == 0) {
#pragma unroll
    for (int r = 0; r < 8; ++r) {
      const int m = hsel * 8 + r;  // C layout: VGPR r, lane-half -> row
      scores[(size_t)b * S_ + s0 + wid * 16 + m] = sp[r];
    }
  }
}

// ---------------- Kernel 3: masked softmax over S per batch row ----------------
__global__ __launch_bounds__(256) void softmax_kernel(const float* __restrict__ scores,
                                                      const unsigned char* __restrict__ mask,
                                                      float* __restrict__ out) {
  const int b = blockIdx.x;
  const int tid = threadIdx.x;
  __shared__ float redmax[8];
  __shared__ float redsum[8];

  float vals[8];
  float mx = -INFINITY;
#pragma unroll
  for (int i = 0; i < 8; ++i) {
    const int idx = i * 256 + tid;
    float s = scores[(size_t)b * S_ + idx];
    s = mask[(size_t)b * S_ + idx] ? s : -INFINITY;
    vals[i] = s;
    mx = fmaxf(mx, s);
  }
  for (int o = 1; o < 32; o <<= 1) mx = fmaxf(mx, __shfl_xor(mx, o, 32));
  if ((tid & 31) == 0) redmax[tid >> 5] = mx;
  __syncthreads();
  mx = redmax[0];
#pragma unroll
  for (int i = 1; i < 8; ++i) mx = fmaxf(mx, redmax[i]);

  float sum = 0.f;
#pragma unroll
  for (int i = 0; i < 8; ++i) {
    vals[i] = __expf(vals[i] - mx);
    sum += vals[i];
  }
  for (int o = 1; o < 32; o <<= 1) sum += __shfl_xor(sum, o, 32);
  if ((tid & 31) == 0) redsum[tid >> 5] = sum;
  __syncthreads();
  float total = 0.f;
#pragma unroll
  for (int i = 0; i < 8; ++i) total += redsum[i];

#pragma unroll
  for (int i = 0; i < 8; ++i) {
    const int idx = i * 256 + tid;
    out[(size_t)b * S_ + idx] = vals[i] / total;
  }
}

// ---------------- Launch ----------------
extern "C" void kernel_launch(void* const* d_in, const int* in_sizes, int n_in,
                              void* d_out, int out_size, void* d_ws, size_t ws_size,
                              hipStream_t stream) {
  const float*         query = (const float*)d_in[0];          // [B, VD]
  const float*         keys  = (const float*)d_in[1];          // [B, S, MD]
  const unsigned char* mask  = (const unsigned char*)d_in[2];  // [B, S] bool
  const float*         W1    = (const float*)d_in[3];          // [AD, VD]
  const float*         W2    = (const float*)d_in[4];          // [AD, MD]
  const float*         v     = (const float*)d_in[5];          // [AD]
  float* out = (float*)d_out;                                  // [B, S]

  float* w1q    = (float*)d_ws;                                // B_*D_ floats
  float* scores = w1q + B_ * D_;                               // B_*S_ floats

  // 1) w1q = query @ W1^T
  w1q_kernel<<<(B_ * D_) / 256, 256, 0, stream>>>(query, W1, w1q);

  // 2) WMMA scores
  dim3 g2(S_ / ROWS, B_);
  size_t smem = (size_t)(ROWS * PITCH + 16 * PITCH) * sizeof(unsigned short)
              + (size_t)2 * D_ * sizeof(float);                // ~150 KB
  att_scores_kernel<<<g2, 256, smem, stream>>>(keys, W2, w1q, v, scores);

  // 3) masked softmax
  softmax_kernel<<<B_, 256, 0, stream>>>(scores, mask, out);
}
